// Recurrence_52329881535122
// MI455X (gfx1250) — matrix operand order repaired and checked by reading, not compiled, not run on hardware
//
#include <hip/hip_runtime.h>
#include <math.h>

#define T_DIM 64
#define N_DIM 1024
#define L_DIM 16
#define H_DIM 256
#define COND_DIM 64
#define NA_DIM 16
#define NL_DIM 64
#define HX_DIM 292
#define G3H 768      // 3*H
#define CIN 576      // COND + 2H

typedef __bf16 bf16_t;
typedef __attribute__((ext_vector_type(16))) __bf16 bf16x16;
typedef __attribute__((ext_vector_type(8)))  __bf16 bf16x8;
typedef __attribute__((ext_vector_type(8)))  float  f32x8;

// ---------------- WMMA helpers (CDNA5 16x16x32 bf16, wave32) ----------------

__device__ __forceinline__ f32x8 wmma_bf16(bf16x16 a, bf16x16 b, f32x8 c) {
  return __builtin_amdgcn_wmma_f32_16x16x32_bf16(false, a, false, b, (short)0, c,
                                                 false, false);
}

// A-matrix 16x32 bf16 fragment from row-major [rows, ld] at (m0, k0).
// ISA layout: lane<16 -> M=lane, K = k0+0..7 then k0+16..23
//             lane>=16 -> M=lane-16, K = k0+8..15 then k0+24..31
__device__ __forceinline__ bf16x16 load_fragA(const bf16_t* base, int ld, int m0, int k0) {
  const int lane = threadIdx.x & 31;
  const int row  = m0 + (lane & 15);
  const int kb   = k0 + ((lane & 16) ? 8 : 0);
  const bf16_t* p = base + (size_t)row * ld + kb;
  bf16x8 lo = *(const bf16x8*)(p);
  bf16x8 hi = *(const bf16x8*)(p + 16);
  bf16x16 a;
#pragma unroll
  for (int i = 0; i < 8; ++i) { a[i] = lo[i]; a[i + 8] = hi[i]; }
  return a;
}

// B-matrix 32x16 bf16 fragment, B[k][n] = W[n0+n][k0+k] with W row-major (Nout, K).
// ISA layout: lane<16 -> N=lane, K = k0+0..7 then k0+16..23 ; lane>=16 -> N=lane-16, K = k0+8..15 then k0+24..31
__device__ __forceinline__ bf16x16 load_fragB(const bf16_t* base, int ld, int n0, int k0) {
  const int lane = threadIdx.x & 31;
  const int row  = n0 + (lane & 15);
  const int kb   = k0 + ((lane & 16) ? 8 : 0);
  const bf16_t* p = base + (size_t)row * ld + kb;
  bf16x8 lo = *(const bf16x8*)(p);
  bf16x8 hi = *(const bf16x8*)(p + 16);
  bf16x16 b;
#pragma unroll
  for (int i = 0; i < 8; ++i) { b[i] = lo[i]; b[i + 8] = hi[i]; }
  return b;
}

// C/D f32 16x16: VGPR r -> rows (m0+r) [lanes 0-15] and (m0+r+8) [lanes 16-31], col n0+lane%16
__device__ __forceinline__ void store_acc(float* base, int ld, int m0, int n0, f32x8 acc) {
  const int lane  = threadIdx.x & 31;
  const int col   = n0 + (lane & 15);
  const int rbase = m0 + ((lane & 16) ? 8 : 0);
#pragma unroll
  for (int r = 0; r < 8; ++r) base[(size_t)(rbase + r) * ld + col] = acc[r];
}

__device__ __forceinline__ void store_acc_bias_global(float* base, int ld, int m0, int n0,
                                                      f32x8 acc, const float* bias) {
  const int lane  = threadIdx.x & 31;
  const int col   = n0 + (lane & 15);
  const int rbase = m0 + ((lane & 16) ? 8 : 0);
  const float b = bias[col];
#pragma unroll
  for (int r = 0; r < 8; ++r) base[(size_t)(rbase + r) * ld + col] = acc[r] + b;
}

__device__ __forceinline__ void store_relu_bf(bf16_t* z, int ld, int n0, f32x8 acc,
                                              const float* bias) {
  const int lane  = threadIdx.x & 31;
  const int col   = n0 + (lane & 15);
  const int rbase = (lane & 16) ? 8 : 0;
  const float b = bias[col];
#pragma unroll
  for (int r = 0; r < 8; ++r)
    z[(size_t)(rbase + r) * ld + col] = (bf16_t)fmaxf(acc[r] + b, 0.0f);
}

__device__ __forceinline__ float sigmoidf(float x) { return 1.0f / (1.0f + expf(-x)); }

// ---------------- small prep kernels ----------------

__global__ void k_f32_to_bf16(const float* __restrict__ src, bf16_t* __restrict__ dst, int n) {
  int i = blockIdx.x * blockDim.x + threadIdx.x;
  if (i < n) dst[i] = (bf16_t)src[i];
}

// gi tables: gi[dir][nl][c] = bih[c] + dot(embed[nl], Wih[c])
__global__ void k_gi_table(const float* __restrict__ embed,
                           const float* __restrict__ WihF, const float* __restrict__ bihF,
                           const float* __restrict__ WihB, const float* __restrict__ bihB,
                           float* __restrict__ giF, float* __restrict__ giB) {
  int i = blockIdx.x * blockDim.x + threadIdx.x;
  if (i >= 2 * NL_DIM * G3H) return;
  int dir = i / (NL_DIM * G3H);
  int r   = i % (NL_DIM * G3H);
  int nl  = r / G3H, c = r % G3H;
  const float* W = dir ? WihB : WihF;
  const float* b = dir ? bihB : bihF;
  const float* x = embed + (size_t)nl * H_DIM;
  const float* w = W + (size_t)c * H_DIM;
  float acc = b[c];
  for (int k = 0; k < H_DIM; ++k) acc += x[k] * w[k];
  (dir ? giB : giF)[(size_t)nl * G3H + c] = acc;
}

__global__ void k_init_h(const float* __restrict__ hx, float* __restrict__ h32,
                         bf16_t* __restrict__ hbf) {
  int i = blockIdx.x * blockDim.x + threadIdx.x;   // N*H
  if (i >= N_DIM * H_DIM) return;
  int n = i >> 8, c = i & 255;
  float v = hx[(size_t)n * HX_DIM + 4 + c];
  h32[i] = v;
  hbf[i] = (bf16_t)v;
}

// ---------------- phase 1: bidirectional GRU over L=16 steps ----------------
// grid: 2048 blocks (dir*1024 + blk), 256 threads, dyn LDS = 73728 B
__global__ void k_phase1(const bf16_t* __restrict__ WhhF, const bf16_t* __restrict__ WhhB,
                         const float* __restrict__ giF, const float* __restrict__ giB,
                         const float* __restrict__ bhhF, const float* __restrict__ bhhB,
                         const int* __restrict__ lines, bf16_t* __restrict__ Gbf) {
  extern __shared__ char smem[];
  float*  h32 = (float*)smem;                                   // 16*256 f32
  bf16_t* hbf = (bf16_t*)(smem + 16 * H_DIM * 4);               // 16*256 bf16
  float*  gh  = (float*)(smem + 16 * H_DIM * 4 + 16 * H_DIM * 2); // 16*768 f32

  const int dir   = blockIdx.x >> 10;
  const int blk   = blockIdx.x & 1023;
  const int j     = blk >> 6;            // shift index (constant per block)
  const int nbase = (blk & 63) * 16;     // batch base
  const bf16_t* Whh = dir ? WhhB : WhhF;
  const float*  giT = dir ? giB : giF;
  const float*  bhh = dir ? bhhB : bhhF;
  const int tid = threadIdx.x;
  const int wave = tid >> 5;

  for (int i = tid; i < 16 * H_DIM; i += blockDim.x) { h32[i] = 0.0f; hbf[i] = (bf16_t)0.0f; }

  for (int k = 0; k < L_DIM; ++k) {
    __syncthreads();
    for (int tile = wave; tile < 48; tile += 8) {
      const int n0 = tile * 16;
      f32x8 acc = {0.f,0.f,0.f,0.f,0.f,0.f,0.f,0.f};
#pragma unroll
      for (int kk = 0; kk < 8; ++kk) {
        bf16x16 a = load_fragA(hbf, H_DIM, 0, kk * 32);
        bf16x16 b = load_fragB(Whh, H_DIM, n0, kk * 32);
        acc = wmma_bf16(a, b, acc);
      }
      store_acc(gh, G3H, 0, n0, acc);
    }
    __syncthreads();
    const int pos = dir ? ((L_DIM - 1 - k + j) & 15) : ((k + j) & 15);
    for (int i = tid; i < 16 * H_DIM; i += blockDim.x) {
      int rr = i >> 8, c = i & 255;
      int n = nbase + rr;
      int line = lines[n * L_DIM + pos];
      const float* gi = giT + (size_t)line * G3H;
      float hr = gh[rr * G3H + c]       + bhh[c];
      float hz = gh[rr * G3H + 256 + c] + bhh[256 + c];
      float hn = gh[rr * G3H + 512 + c] + bhh[512 + c];
      float r  = sigmoidf(gi[c] + hr);
      float z  = sigmoidf(gi[256 + c] + hz);
      float nn = tanhf(gi[512 + c] + r * hn);
      float h = h32[i];
      h = (1.0f - z) * nn + z * h;
      h32[i] = h;
      hbf[i] = (bf16_t)h;
    }
  }
  __syncthreads();
  for (int i = tid; i < 16 * H_DIM; i += blockDim.x) {
    int rr = i >> 8, c = i & 255;
    int n = nbase + rr;
    Gbf[((size_t)(j * N_DIM + n)) * 512 + dir * 256 + c] = (bf16_t)h32[i];
  }
}

// ---------------- Ggi = G @ Wih_c[:,64:].T + bih_c  (16384 x 768) ----------------
__global__ void k_ggi(const bf16_t* __restrict__ Gbf, const bf16_t* __restrict__ WihC,
                      const float* __restrict__ bihC, float* __restrict__ Ggi) {
  const int m0 = blockIdx.x * 16;
  const int wave = threadIdx.x >> 5;
  for (int tile = wave; tile < 48; tile += 8) {
    const int n0 = tile * 16;
    f32x8 acc = {0.f,0.f,0.f,0.f,0.f,0.f,0.f,0.f};
#pragma unroll
    for (int kk = 0; kk < 16; ++kk) {
      bf16x16 a = load_fragA(Gbf, 512, m0, kk * 32);
      bf16x16 b = load_fragB(WihC + COND_DIM, CIN, n0, kk * 32);  // columns 64..575
      acc = wmma_bf16(a, b, acc);
    }
    store_acc_bias_global(Ggi, G3H, m0, n0, acc, bihC);
  }
}

// ---------------- phase 2 step, part 1: GRU cell update ----------------
// grid: 64 blocks, 256 threads, dyn LDS = 98304 B
__global__ void k_d1(int t, const bf16_t* __restrict__ condbf, const bf16_t* __restrict__ WihC,
                     const bf16_t* __restrict__ WhhC, const float* __restrict__ Ggi,
                     const float* __restrict__ bhhC, const int* __restrict__ active,
                     float* __restrict__ h32g, bf16_t* __restrict__ hbfg) {
  extern __shared__ char smem[];
  float* ci = (float*)smem;          // 16*768 input-gate accum (cond part)
  float* ch = ci + 16 * G3H;         // 16*768 hidden-gate accum

  const int n0 = blockIdx.x * 16;
  const int wave = threadIdx.x >> 5;
  const bf16_t* condrow = condbf + (size_t)t * N_DIM * COND_DIM;

  for (int tile = wave; tile < 48; tile += 8) {
    const int c0 = tile * 16;
    f32x8 aci = {0.f,0.f,0.f,0.f,0.f,0.f,0.f,0.f};
    f32x8 ach = {0.f,0.f,0.f,0.f,0.f,0.f,0.f,0.f};
#pragma unroll
    for (int kk = 0; kk < 2; ++kk) {   // cond part, K=64
      bf16x16 a = load_fragA(condrow, COND_DIM, n0, kk * 32);
      bf16x16 b = load_fragB(WihC, CIN, c0, kk * 32);
      aci = wmma_bf16(a, b, aci);
    }
#pragma unroll
    for (int kk = 0; kk < 8; ++kk) {   // hidden part, K=256
      bf16x16 a = load_fragA(hbfg, H_DIM, n0, kk * 32);
      bf16x16 b = load_fragB(WhhC, H_DIM, c0, kk * 32);
      ach = wmma_bf16(a, b, ach);
    }
    store_acc(ci, G3H, 0, c0, aci);
    store_acc(ch, G3H, 0, c0, ach);
  }
  __syncthreads();
  for (int i = threadIdx.x; i < 16 * H_DIM; i += blockDim.x) {
    int rr = i >> 8, c = i & 255;
    int n = n0 + rr;
    int l = active[t * N_DIM + n];
    const float* grow = Ggi + ((size_t)l * N_DIM + n) * G3H;   // gathered g-part (+bih)
    float ir  = ci[rr * G3H + c]       + grow[c];
    float iz  = ci[rr * G3H + 256 + c] + grow[256 + c];
    float inn = ci[rr * G3H + 512 + c] + grow[512 + c];
    float hr  = ch[rr * G3H + c]       + bhhC[c];
    float hz  = ch[rr * G3H + 256 + c] + bhhC[256 + c];
    float hn  = ch[rr * G3H + 512 + c] + bhhC[512 + c];
    float r  = sigmoidf(ir + hr);
    float z  = sigmoidf(iz + hz);
    float nn = tanhf(inn + r * hn);
    float h = h32g[(size_t)n * H_DIM + c];
    h = (1.0f - z) * nn + z * h;
    h32g[(size_t)n * H_DIM + c] = h;
    hbfg[(size_t)n * H_DIM + c] = (bf16_t)h;
  }
}

// ---------------- phase 2 step, part 2: MLP + heads + output ----------------
__global__ void k_d2(int t, const bf16_t* __restrict__ hbfg, const float* __restrict__ h32g,
                     const bf16_t* __restrict__ W0b, const bf16_t* __restrict__ W1b,
                     const bf16_t* __restrict__ W2b, const bf16_t* __restrict__ Wab,
                     const float* __restrict__ b0, const float* __restrict__ b1,
                     const float* __restrict__ b2, const float* __restrict__ ba,
                     const float* __restrict__ Wc, const float* __restrict__ bc,
                     const int* __restrict__ active, const int* __restrict__ actions,
                     const float* __restrict__ hx, float* __restrict__ out) {
  __shared__ bf16_t zA[16 * 256];
  __shared__ bf16_t zB[16 * 256];
  __shared__ float  lg[16 * 16];
  __shared__ float  vbuf[16];

  const int n0 = blockIdx.x * 16;
  const int wave = threadIdx.x >> 5;
  const int lane = threadIdx.x & 31;

  // layer 0: h -> zA
  for (int tile = wave; tile < 16; tile += 8) {
    const int c0 = tile * 16;
    f32x8 acc = {0.f,0.f,0.f,0.f,0.f,0.f,0.f,0.f};
#pragma unroll
    for (int kk = 0; kk < 8; ++kk)
      acc = wmma_bf16(load_fragA(hbfg, H_DIM, n0, kk * 32),
                      load_fragB(W0b, H_DIM, c0, kk * 32), acc);
    store_relu_bf(zA, 256, c0, acc, b0);
  }
  __syncthreads();
  // layer 1: zA -> zB
  for (int tile = wave; tile < 16; tile += 8) {
    const int c0 = tile * 16;
    f32x8 acc = {0.f,0.f,0.f,0.f,0.f,0.f,0.f,0.f};
#pragma unroll
    for (int kk = 0; kk < 8; ++kk)
      acc = wmma_bf16(load_fragA(zA, 256, 0, kk * 32),
                      load_fragB(W1b, H_DIM, c0, kk * 32), acc);
    store_relu_bf(zB, 256, c0, acc, b1);
  }
  __syncthreads();
  // layer 2: zB -> zA
  for (int tile = wave; tile < 16; tile += 8) {
    const int c0 = tile * 16;
    f32x8 acc = {0.f,0.f,0.f,0.f,0.f,0.f,0.f,0.f};
#pragma unroll
    for (int kk = 0; kk < 8; ++kk)
      acc = wmma_bf16(load_fragA(zB, 256, 0, kk * 32),
                      load_fragB(W2b, H_DIM, c0, kk * 32), acc);
    store_relu_bf(zA, 256, c0, acc, b2);
  }
  __syncthreads();
  // logits (16 cols): wave 0 only
  if (wave == 0) {
    f32x8 acc = {0.f,0.f,0.f,0.f,0.f,0.f,0.f,0.f};
#pragma unroll
    for (int kk = 0; kk < 8; ++kk)
      acc = wmma_bf16(load_fragA(zA, 256, 0, kk * 32),
                      load_fragB(Wab, H_DIM, 0, kk * 32), acc);
    const int col = lane & 15;
    const int rbase = (lane & 16) ? 8 : 0;
    const float bias = ba[col];
#pragma unroll
    for (int r = 0; r < 8; ++r) lg[(rbase + r) * 16 + col] = acc[r] + bias;
  }
  __syncthreads();
  // value + softmax, one thread per row
  if (threadIdx.x < 16) {
    const int rr = threadIdx.x;
    float v = bc[0];
    for (int k = 0; k < H_DIM; ++k) v += (float)zA[rr * 256 + k] * Wc[k];
    vbuf[rr] = v;
    float m = lg[rr * 16];
    for (int k2 = 1; k2 < 16; ++k2) m = fmaxf(m, lg[rr * 16 + k2]);
    float e[16], s = 0.0f;
    for (int k2 = 0; k2 < 16; ++k2) { e[k2] = expf(lg[rr * 16 + k2] - m); s += e[k2]; }
    float inv = 1.0f / s;
    for (int k2 = 0; k2 < 16; ++k2) lg[rr * 16 + k2] = e[k2] * inv;
  }
  __syncthreads();
  // emit output channels [A,P,Wo,V,H(256),probs(16),p0(16)]
  for (int i = threadIdx.x; i < 16 * HX_DIM; i += blockDim.x) {
    int rr = i / HX_DIM, c = i % HX_DIM;
    int n = n0 + rr;
    float val;
    if (c == 0)        val = (float)actions[(t * N_DIM + n) * 2 + 0];  // sample never used
    else if (c == 1)   val = (float)actions[(t * N_DIM + n) * 2 + 1];
    else if (c == 2)   val = (float)active[t * N_DIM + n];
    else if (c == 3)   val = vbuf[rr];
    else if (c < 260)  val = h32g[(size_t)n * H_DIM + (c - 4)];
    else if (c < 276)  val = lg[rr * 16 + (c - 260)];
    else               val = hx[(size_t)n * HX_DIM + c];               // p_probs0 passthrough
    out[((size_t)t * N_DIM + n) * HX_DIM + c] = val;
  }
}

// ---------------- host launcher ----------------

extern "C" void kernel_launch(void* const* d_in, const int* in_sizes, int n_in,
                              void* d_out, int out_size, void* d_ws, size_t ws_size,
                              hipStream_t stream) {
  (void)in_sizes; (void)n_in; (void)out_size; (void)ws_size;
  const float* condition = (const float*)d_in[0];
  const float* hx        = (const float*)d_in[1];
  const float* embed     = (const float*)d_in[2];
  const float* Wih_f = (const float*)d_in[3];
  const float* Whh_f = (const float*)d_in[4];
  const float* bih_f = (const float*)d_in[5];
  const float* bhh_f = (const float*)d_in[6];
  const float* Wih_b = (const float*)d_in[7];
  const float* Whh_b = (const float*)d_in[8];
  const float* bih_b = (const float*)d_in[9];
  const float* bhh_b = (const float*)d_in[10];
  const float* Wih_c = (const float*)d_in[11];
  const float* Whh_c = (const float*)d_in[12];
  const float* bih_c = (const float*)d_in[13];
  const float* bhh_c = (const float*)d_in[14];
  const float* W0 = (const float*)d_in[15];
  const float* b0 = (const float*)d_in[16];
  const float* W1 = (const float*)d_in[17];
  const float* b1 = (const float*)d_in[18];
  const float* W2 = (const float*)d_in[19];
  const float* b2 = (const float*)d_in[20];
  const float* Wc = (const float*)d_in[21];
  const float* bc = (const float*)d_in[22];
  const float* Wa = (const float*)d_in[23];
  const float* ba = (const float*)d_in[24];
  const int* lines   = (const int*)d_in[25];
  const int* active  = (const int*)d_in[26];
  const int* actions = (const int*)d_in[27];
  float* out = (float*)d_out;

  // workspace layout (bytes)
  char* ws = (char*)d_ws;
  bf16_t* WhhF_bf = (bf16_t*)(ws + 0);                 // 768*256
  bf16_t* WhhB_bf = (bf16_t*)(ws + 393216);
  bf16_t* WhhC_bf = (bf16_t*)(ws + 786432);
  bf16_t* WihC_bf = (bf16_t*)(ws + 1179648);           // 768*576
  bf16_t* W0_bf   = (bf16_t*)(ws + 2064384);           // 256*256
  bf16_t* W1_bf   = (bf16_t*)(ws + 2195456);
  bf16_t* W2_bf   = (bf16_t*)(ws + 2326528);
  bf16_t* Wa_bf   = (bf16_t*)(ws + 2457600);           // 16*256
  bf16_t* cond_bf = (bf16_t*)(ws + 2465792);           // T*N*64
  float*  giF     = (float*) (ws + 10854400);          // 64*768
  float*  giB     = (float*) (ws + 11051008);
  bf16_t* Gbf     = (bf16_t*)(ws + 11247616);          // 16384*512
  float*  Ggi     = (float*) (ws + 28024832);          // 16384*768
  float*  h32g    = (float*) (ws + 78356480);          // 1024*256
  bf16_t* hbfg    = (bf16_t*)(ws + 79405056);          // 1024*256

  auto conv = [&](const float* s, bf16_t* d, int n) {
    k_f32_to_bf16<<<(n + 255) / 256, 256, 0, stream>>>(s, d, n);
  };
  conv(Whh_f, WhhF_bf, G3H * H_DIM);
  conv(Whh_b, WhhB_bf, G3H * H_DIM);
  conv(Whh_c, WhhC_bf, G3H * H_DIM);
  conv(Wih_c, WihC_bf, G3H * CIN);
  conv(W0, W0_bf, H_DIM * H_DIM);
  conv(W1, W1_bf, H_DIM * H_DIM);
  conv(W2, W2_bf, H_DIM * H_DIM);
  conv(Wa, Wa_bf, NA_DIM * H_DIM);
  conv(condition, cond_bf, T_DIM * N_DIM * COND_DIM);

  k_gi_table<<<(2 * NL_DIM * G3H + 255) / 256, 256, 0, stream>>>(
      embed, Wih_f, bih_f, Wih_b, bih_b, giF, giB);
  k_init_h<<<(N_DIM * H_DIM) / 256, 256, 0, stream>>>(hx, h32g, hbfg);

  // phase 1: both directions, 16 steps, 16384-row batch
  k_phase1<<<2048, 256, 73728, stream>>>(WhhF_bf, WhhB_bf, giF, giB, bhh_f, bhh_b,
                                         lines, Gbf);
  // precompute g-part of input gates for all (l, n)
  k_ggi<<<1024, 256, 0, stream>>>(Gbf, WihC_bf, bih_c, Ggi);

  // phase 2: 64 sequential steps (stream-ordered)
  for (int t = 0; t < T_DIM; ++t) {
    k_d1<<<64, 256, 98304, stream>>>(t, cond_bf, WihC_bf, WhhC_bf, Ggi, bhh_c,
                                     active, h32g, hbfg);
    k_d2<<<64, 256, 0, stream>>>(t, hbfg, h32g, W0_bf, W1_bf, W2_bf, Wa_bf,
                                 b0, b1, b2, ba, Wc, bc, active, actions, hx, out);
  }
}
